// HierarchicalWTABlockV3_27144193310738
// MI455X (gfx1250) — matrix operand order repaired and checked by reading, not compiled, not run on hardware
//
#include <hip/hip_runtime.h>
#include <hip/hip_bf16.h>
#include <math.h>

// ---------------------------------------------------------------------------
// Problem constants (match reference)
// ---------------------------------------------------------------------------
#define BB 8
#define LL 4096
#define GG 64
#define KK 16
#define DD 64
#define NN (GG * KK)          // 1024
#define TWO_D (2 * DD)        // 128
#define EPSV 1e-5f

typedef __attribute__((ext_vector_type(16))) _Float16 v16h;
typedef __attribute__((ext_vector_type(8)))  float    v8f;
typedef int v4i_t __attribute__((vector_size(16)));   // <4 x i32>

union AFrag { v16h v; _Float16 h[16]; };
union CFrag { v8f  v; float    f[8];  };

__device__ inline v8f wmma_f16f32(v16h a, v16h b, v8f c) {
  // D = A(16x32 f16) * B(32x16 f16) + C(16x16 f32)
  return __builtin_amdgcn_wmma_f32_16x16x32_f16(
      /*neg_a=*/false, a, /*neg_b=*/false, b,
      /*c_mod=*/(short)0, c, /*reuse_a=*/false, /*reuse_b=*/false);
}

__device__ inline float gelu_exact(float x) {
  return 0.5f * x * (1.0f + erff(x * 0.70710678118654752440f));
}
__device__ inline float sigmoidf(float x) {
  return 1.0f / (1.0f + expf(-x));
}

// ---------------------------------------------------------------------------
// Async global->LDS staging (CDNA5 GLOBAL_LOAD_ASYNC_TO_LDS_B128, ASYNCcnt)
// Builtin signature (from toolchain diagnostic): (v4i32 as1*, v4i32 as3*, Ii, Ii)
// ---------------------------------------------------------------------------
#if __has_builtin(__builtin_amdgcn_global_load_async_to_lds_b128)
#define HAVE_ASYNC_LDS 1
__device__ inline void async_copy16(const void* g, void* l) {
  __builtin_amdgcn_global_load_async_to_lds_b128(
      (__attribute__((address_space(1))) v4i_t*)(void*)g,
      (__attribute__((address_space(3))) v4i_t*)l, 0, 0);
}
#else
#define HAVE_ASYNC_LDS 0
#endif

__device__ inline void wait_async_zero() {
#if __has_builtin(__builtin_amdgcn_s_wait_asynccnt)
  __builtin_amdgcn_s_wait_asynccnt(0);
#else
  asm volatile("s_wait_asynccnt 0" ::: "memory");
#endif
}

// A-fragment gather (16x32 f16 tile of a row-major f16 LDS tile, stride ld).
// CDNA5 ISA: lanes 0-15 hold M=lane; lanes 16-31 hold M=lane-16 with the
// other K sub-band.  Halfword e -> VGPR j=e/2, pos=e%2,
// k = (j>=4?16:0) + half*8 + 2*(j&3) + pos  within the 32-wide K tile.
__device__ inline AFrag gather_a(const _Float16* t, int ld, int lane, int kt) {
  AFrag a;
  const int m = lane & 15;
  const int half = (lane >> 4) & 1;
#pragma unroll
  for (int e = 0; e < 16; ++e) {
    const int j = e >> 1, pos = e & 1;
    const int k = kt * 32 + ((j >= 4) ? 16 : 0) + half * 8 + ((j & 3) << 1) + pos;
    a.h[e] = t[m * ld + k];
  }
  return a;
}

// ---------------------------------------------------------------------------
// Prep kernels
// ---------------------------------------------------------------------------
__global__ void zero_f32_kernel(float* p, int n) {
  for (int i = blockIdx.x * blockDim.x + threadIdx.x; i < n; i += gridDim.x * blockDim.x)
    p[i] = 0.0f;
}

__global__ void ssum_kernel(const float* __restrict__ S, float* __restrict__ ssum) {
  const int b = blockIdx.x;        // 0..B-1
  const int d = threadIdx.x;       // 0..63
  float acc = 0.0f;
  for (int n = 0; n < NN; ++n)
    acc += S[((size_t)b * NN + n) * DD + d];
  ssum[b * DD + d] = acc * (1.0f / (float)NN);
}

// XSh[row][0:64) = X[row], XSh[row][64:128) = S_sum[batch(row)]  (f16)
__global__ void build_xsh_kernel(const float* __restrict__ X,
                                 const float* __restrict__ ssum,
                                 _Float16* __restrict__ XSh) {
  for (int i = blockIdx.x * blockDim.x + threadIdx.x; i < BB * LL * TWO_D;
       i += gridDim.x * blockDim.x) {
    const int row = i >> 7, c = i & 127;
    const int b = row >> 12;  // / L
    const float v = (c < DD) ? X[(size_t)row * DD + c] : ssum[b * DD + (c - DD)];
    XSh[i] = (_Float16)v;
  }
}

// UIh[row][0:64) = S[row], UIh[row][64:128) = incoming[row]  (f16)
__global__ void build_uih_kernel(const float* __restrict__ S,
                                 const float* __restrict__ incoming,
                                 _Float16* __restrict__ UIh) {
  for (int i = blockIdx.x * blockDim.x + threadIdx.x; i < BB * NN * TWO_D;
       i += gridDim.x * blockDim.x) {
    const int row = i >> 7, c = i & 127;
    const float v = (c < DD) ? S[(size_t)row * DD + c]
                             : incoming[(size_t)row * DD + (c - DD)];
    UIh[i] = (_Float16)v;
  }
}

// Pack weight W[O x 128] (row-major, f32) into WMMA B-fragments (f16).
// Fragment f = ct*4 + kt covers out-cols [ct*16,ct*16+16) x K [kt*32,kt*32+32).
// B layout (symmetric to ISA SWMMAC B description): lane holds column
// n = ct*16 + (lane&15); halfword e -> k = kt*32 + (e>=8?16:0) + (lane>=16?8:0) + (e&7).
__global__ void pack_w_kernel(const float* __restrict__ W, _Float16* __restrict__ dst) {
  const int f = blockIdx.x;
  const int ct = f >> 2, kt = f & 3;
  const int lane = threadIdx.x;            // 0..31
  const int half = lane >> 4;
  const int n = ct * 16 + (lane & 15);
#pragma unroll
  for (int e = 0; e < 16; ++e) {
    const int k = kt * 32 + ((e >= 8) ? 16 : 0) + half * 8 + (e & 7);
    dst[(size_t)f * 512 + lane * 16 + e] = (_Float16)W[n * TWO_D + k];
  }
}

// ---------------------------------------------------------------------------
// Tile staging: DMA one 16x128 f16 tile (row-major, global) into a padded
// LDS tile (stride LDP halfwords).  16 rows x 16 chunks of 16B = 256 chunks.
// ---------------------------------------------------------------------------
#define LDP 136   // padded LDS stride in halfwords (272B, keeps 16B alignment)

__device__ inline void stage_tile(const _Float16* __restrict__ g_row0,
                                  _Float16* __restrict__ lds_tile, int lane) {
#if HAVE_ASYNC_LDS
#pragma unroll
  for (int it = 0; it < 8; ++it) {
    const int q = it * 32 + lane;          // chunk id
    const int r = q >> 4, ch = q & 15;     // row, 16B-chunk within row
    async_copy16(g_row0 + (size_t)r * TWO_D + ch * 8,
                 lds_tile + r * LDP + ch * 8);
  }
  wait_async_zero();
#else
#pragma unroll
  for (int it = 0; it < 8; ++it) {
    const int q = it * 32 + lane;
    const int r = q >> 4, ch = q & 15;
    const uint4 v = *(const uint4*)(g_row0 + (size_t)r * TWO_D + ch * 8);
    *(uint4*)(lds_tile + r * LDP + ch * 8) = v;
  }
#endif
}

// ---------------------------------------------------------------------------
// Token kernel: msg/gate MLPs + WTA routing + scatter-add into `incoming`.
// 4 wave32s per block, one 16-token tile per wave.
// ---------------------------------------------------------------------------
#define TW 4  // waves per block

__global__ __launch_bounds__(32 * TW) void tok_kernel(
    const _Float16* __restrict__ XSh,
    const _Float16* __restrict__ pmw0, const float* __restrict__ mb0,
    const _Float16* __restrict__ pmw1, const float* __restrict__ mb1,
    const _Float16* __restrict__ pgw0, const float* __restrict__ gb0,
    const float* __restrict__ gw1, const float* __restrict__ gb1,
    const _Float16* __restrict__ pwg, const _Float16* __restrict__ pws,
    float* __restrict__ incoming) {
  __shared__ __align__(16) _Float16 xs[TW][16][LDP];  // XS tile, f16
  __shared__ __align__(16) _Float16 h1[TW][16][132];  // hidden of msg MLP
  __shared__ float    sc[TW][16][68];                 // f32 scratch
  __shared__ float    gate_s[TW][16];
  __shared__ int      slot_s[TW][16];

  const int lane = threadIdx.x & 31;
  const int w    = threadIdx.x >> 5;
  const int tile = blockIdx.x * TW + w;
  const int row0 = tile * 16;            // flat token index (B*L)
  const int b    = row0 >> 12;           // / 4096

  const v16h* bmw0 = (const v16h*)pmw0;
  const v16h* bmw1 = (const v16h*)pmw1;
  const v16h* bgw0 = (const v16h*)pgw0;
  const v16h* bwg  = (const v16h*)pwg;
  const v16h* bws  = (const v16h*)pws;

  __builtin_prefetch(pmw0, 0, 1);
  __builtin_prefetch(pgw0, 0, 1);
  __builtin_prefetch(pwg, 0, 1);

  // ---- DMA-stage XS tile (f16, pre-built) --------------------------------
  stage_tile(XSh + (size_t)row0 * TWO_D, &xs[w][0][0], lane);
  __syncthreads();

  AFrag a[4];
#pragma unroll
  for (int kt = 0; kt < 4; ++kt) a[kt] = gather_a(&xs[w][0][0], LDP, lane, kt);

  const int ncol = lane & 15;
  const int half = lane >> 4;

  // ---- hidden1 = gelu(XS @ mw0.T + mb0)  [16 x 128] ----------------------
#pragma unroll
  for (int ct = 0; ct < 8; ++ct) {
    CFrag c = {};
#pragma unroll
    for (int kt = 0; kt < 4; ++kt)
      c.v = wmma_f16f32(a[kt].v, bmw0[(ct * 4 + kt) * 32 + lane], c.v);
    const float bias = mb0[ct * 16 + ncol];
#pragma unroll
    for (int r = 0; r < 8; ++r)
      h1[w][half * 8 + r][ct * 16 + ncol] = (_Float16)gelu_exact(c.f[r] + bias);
  }

  // ---- gate hidden = gelu(XS @ gw0.T + gb0)  [16 x 64] -> sc -------------
#pragma unroll
  for (int ct = 0; ct < 4; ++ct) {
    CFrag c = {};
#pragma unroll
    for (int kt = 0; kt < 4; ++kt)
      c.v = wmma_f16f32(a[kt].v, bgw0[(ct * 4 + kt) * 32 + lane], c.v);
    const float bias = gb0[ct * 16 + ncol];
#pragma unroll
    for (int r = 0; r < 8; ++r)
      sc[w][half * 8 + r][ct * 16 + ncol] = gelu_exact(c.f[r] + bias);
  }
  __syncthreads();

  // ---- gate = sigmoid(ghid . gw1 + gb1) ----------------------------------
  if (lane < 16) {
    float acc = gb1[0];
    for (int c = 0; c < DD; ++c) acc += sc[w][lane][c] * gw1[c];
    gate_s[w][lane] = sigmoidf(acc);
  }
  __syncthreads();

  // ---- group logits = XS @ Wg.T  [16 x 64] -> sc, argmax -----------------
#pragma unroll
  for (int ct = 0; ct < 4; ++ct) {
    CFrag c = {};
#pragma unroll
    for (int kt = 0; kt < 4; ++kt)
      c.v = wmma_f16f32(a[kt].v, bwg[(ct * 4 + kt) * 32 + lane], c.v);
#pragma unroll
    for (int r = 0; r < 8; ++r)
      sc[w][half * 8 + r][ct * 16 + ncol] = c.f[r];
  }
  __syncthreads();
  if (lane < 16) {
    float best = sc[w][lane][0];
    int bi = 0;
    for (int c = 1; c < GG; ++c) {
      const float v = sc[w][lane][c];
      if (v > best) { best = v; bi = c; }
    }
    slot_s[w][lane] = bi * KK;           // g_idx * K
  }
  __syncthreads();

  // ---- slot logits = XS @ Ws.T  [16 x 16] -> sc, argmax ------------------
  {
    CFrag c = {};
#pragma unroll
    for (int kt = 0; kt < 4; ++kt)
      c.v = wmma_f16f32(a[kt].v, bws[kt * 32 + lane], c.v);
#pragma unroll
    for (int r = 0; r < 8; ++r)
      sc[w][half * 8 + r][ncol] = c.f[r];
  }
  __syncthreads();
  if (lane < 16) {
    float best = sc[w][lane][0];
    int bi = 0;
    for (int c = 1; c < KK; ++c) {
      const float v = sc[w][lane][c];
      if (v > best) { best = v; bi = c; }
    }
    slot_s[w][lane] += bi;               // full flat slot in [0,N)
  }
  __syncthreads();

  // ---- msg = hidden1 @ mw1.T + mb1  [16 x 64] -> sc ----------------------
  AFrag ah[4];
#pragma unroll
  for (int kt = 0; kt < 4; ++kt) ah[kt] = gather_a(&h1[w][0][0], 132, lane, kt);
#pragma unroll
  for (int ct = 0; ct < 4; ++ct) {
    CFrag c = {};
#pragma unroll
    for (int kt = 0; kt < 4; ++kt)
      c.v = wmma_f16f32(ah[kt].v, bmw1[(ct * 4 + kt) * 32 + lane], c.v);
    const float bias = mb1[ct * 16 + ncol];
#pragma unroll
    for (int r = 0; r < 8; ++r)
      sc[w][half * 8 + r][ct * 16 + ncol] = c.f[r] + bias;
  }
  __syncthreads();

  // ---- scatter-add gated messages into incoming[b, slot, :] --------------
  for (int r = 0; r < 16; ++r) {
    const int   slot = slot_s[w][r];
    const float g    = gate_s[w][r];
    float* dst = &incoming[((size_t)b * NN + slot) * DD];
    for (int c = lane; c < DD; c += 32)
      atomicAdd(&dst[c], sc[w][r][c] * g);
  }
}

// ---------------------------------------------------------------------------
// Update kernel: S_new = S + MLP([S|incoming]); LayerNorm -> out
// ---------------------------------------------------------------------------
__global__ __launch_bounds__(32 * TW) void upd_kernel(
    const float* __restrict__ S, const _Float16* __restrict__ UIh,
    const _Float16* __restrict__ puw0, const float* __restrict__ ub0,
    const _Float16* __restrict__ puw1, const float* __restrict__ ub1,
    const float* __restrict__ lnw, const float* __restrict__ lnb,
    float* __restrict__ out) {
  __shared__ __align__(16) _Float16 ui[TW][16][LDP];  // [S | incoming] tile
  __shared__ __align__(16) _Float16 hh[TW][16][132];  // hidden, f16
  __shared__ float    sc[TW][16][68];                 // S_new tile, f32

  const int lane = threadIdx.x & 31;
  const int w    = threadIdx.x >> 5;
  const int tile = blockIdx.x * TW + w;
  const int row0 = tile * 16;            // flat row in [0, B*N)

  const v16h* buw0 = (const v16h*)puw0;
  const v16h* buw1 = (const v16h*)puw1;

  __builtin_prefetch(puw0, 0, 1);
  __builtin_prefetch(puw1, 0, 1);

  stage_tile(UIh + (size_t)row0 * TWO_D, &ui[w][0][0], lane);
  __syncthreads();

  AFrag a[4];
#pragma unroll
  for (int kt = 0; kt < 4; ++kt) a[kt] = gather_a(&ui[w][0][0], LDP, lane, kt);

  const int ncol = lane & 15;
  const int half = lane >> 4;

  // hidden = gelu(upd_in @ uw0.T + ub0)  [16 x 128]
#pragma unroll
  for (int ct = 0; ct < 8; ++ct) {
    CFrag c = {};
#pragma unroll
    for (int kt = 0; kt < 4; ++kt)
      c.v = wmma_f16f32(a[kt].v, buw0[(ct * 4 + kt) * 32 + lane], c.v);
    const float bias = ub0[ct * 16 + ncol];
#pragma unroll
    for (int r = 0; r < 8; ++r)
      hh[w][half * 8 + r][ct * 16 + ncol] = (_Float16)gelu_exact(c.f[r] + bias);
  }
  __syncthreads();

  AFrag ahh[4];
#pragma unroll
  for (int kt = 0; kt < 4; ++kt) ahh[kt] = gather_a(&hh[w][0][0], 132, lane, kt);

  // S_new = S + hidden @ uw1.T + ub1  [16 x 64]
#pragma unroll
  for (int ct = 0; ct < 4; ++ct) {
    CFrag c = {};
#pragma unroll
    for (int kt = 0; kt < 4; ++kt)
      c.v = wmma_f16f32(ahh[kt].v, buw1[(ct * 4 + kt) * 32 + lane], c.v);
    const float bias = ub1[ct * 16 + ncol];
#pragma unroll
    for (int r = 0; r < 8; ++r) {
      const int m = half * 8 + r;
      const int n = ct * 16 + ncol;
      sc[w][m][n] = S[(size_t)(row0 + m) * DD + n] + c.f[r] + bias;
    }
  }
  __syncthreads();

  // LayerNorm per row (lanes 0..15, one row each)
  if (lane < 16) {
    const int r = lane;
    float mu = 0.0f;
    for (int c = 0; c < DD; ++c) mu += sc[w][r][c];
    mu *= (1.0f / (float)DD);
    float var = 0.0f;
    for (int c = 0; c < DD; ++c) {
      const float d = sc[w][r][c] - mu;
      var += d * d;
    }
    var *= (1.0f / (float)DD);
    const float inv = rsqrtf(var + EPSV);
    float* o = &out[(size_t)(row0 + r) * DD];
    for (int c = 0; c < DD; ++c)
      o[c] = (sc[w][r][c] - mu) * inv * lnw[c] + lnb[c];
  }
}

// ---------------------------------------------------------------------------
// Host launcher
// ---------------------------------------------------------------------------
static inline size_t align_up(size_t x, size_t a) { return (x + a - 1) & ~(a - 1); }

extern "C" void kernel_launch(void* const* d_in, const int* in_sizes, int n_in,
                              void* d_out, int out_size, void* d_ws, size_t ws_size,
                              hipStream_t stream) {
  (void)in_sizes; (void)n_in; (void)out_size; (void)ws_size;
  const float* X   = (const float*)d_in[0];
  const float* S   = (const float*)d_in[1];
  const float* Wg  = (const float*)d_in[2];
  const float* Ws  = (const float*)d_in[3];
  const float* mw0 = (const float*)d_in[4];
  const float* mb0 = (const float*)d_in[5];
  const float* mw1 = (const float*)d_in[6];
  const float* mb1 = (const float*)d_in[7];
  const float* gw0 = (const float*)d_in[8];
  const float* gb0 = (const float*)d_in[9];
  const float* gw1 = (const float*)d_in[10];
  const float* gb1 = (const float*)d_in[11];
  const float* uw0 = (const float*)d_in[12];
  const float* ub0 = (const float*)d_in[13];
  const float* uw1 = (const float*)d_in[14];
  const float* ub1 = (const float*)d_in[15];
  const float* lnw = (const float*)d_in[16];
  const float* lnb = (const float*)d_in[17];
  float* out = (float*)d_out;

  char* base = (char*)d_ws;
  size_t off = 0;
  auto alloc = [&](size_t bytes) -> void* {
    off = align_up(off, 256);
    void* p = base + off;
    off += bytes;
    return p;
  };
  float*     ssum     = (float*)alloc((size_t)BB * DD * sizeof(float));
  float*     incoming = (float*)alloc((size_t)BB * NN * DD * sizeof(float));
  _Float16*  XSh      = (_Float16*)alloc((size_t)BB * LL * TWO_D * sizeof(_Float16));
  _Float16*  UIh      = (_Float16*)alloc((size_t)BB * NN * TWO_D * sizeof(_Float16));
  _Float16*  pmw0     = (_Float16*)alloc((size_t)32 * 512 * sizeof(_Float16));
  _Float16*  pmw1     = (_Float16*)alloc((size_t)16 * 512 * sizeof(_Float16));
  _Float16*  pgw0     = (_Float16*)alloc((size_t)16 * 512 * sizeof(_Float16));
  _Float16*  pwg      = (_Float16*)alloc((size_t)16 * 512 * sizeof(_Float16));
  _Float16*  pws      = (_Float16*)alloc((size_t)4  * 512 * sizeof(_Float16));
  _Float16*  puw0     = (_Float16*)alloc((size_t)32 * 512 * sizeof(_Float16));
  _Float16*  puw1     = (_Float16*)alloc((size_t)16 * 512 * sizeof(_Float16));

  // 1) zero scatter buffer (deterministic: re-zero every call)
  zero_f32_kernel<<<1024, 256, 0, stream>>>(incoming, BB * NN * DD);
  // 2) state summary, then pre-built f16 activation matrix for token path
  ssum_kernel<<<BB, DD, 0, stream>>>(S, ssum);
  build_xsh_kernel<<<2048, 256, 0, stream>>>(X, ssum, XSh);
  // 3) pack all weight matrices into WMMA B-fragment layout (f16)
  pack_w_kernel<<<32, 32, 0, stream>>>(mw0, pmw0);
  pack_w_kernel<<<16, 32, 0, stream>>>(mw1, pmw1);
  pack_w_kernel<<<16, 32, 0, stream>>>(gw0, pgw0);
  pack_w_kernel<<<16, 32, 0, stream>>>(Wg,  pwg);
  pack_w_kernel<<<4,  32, 0, stream>>>(Ws,  pws);
  pack_w_kernel<<<32, 32, 0, stream>>>(uw0, puw0);
  pack_w_kernel<<<16, 32, 0, stream>>>(uw1, puw1);
  // 4) token pipeline: 32768 tokens / 16 per tile / 4 tiles per block
  tok_kernel<<<(BB * LL) / 16 / TW, 32 * TW, 0, stream>>>(
      XSh, pmw0, mb0, pmw1, mb1, pgw0, gb0, gw1, gb1, pwg, pws, incoming);
  // 5) build f16 [S|incoming], then slot update + LayerNorm: 8192 rows
  build_uih_kernel<<<1024, 256, 0, stream>>>(S, incoming, UIh);
  upd_kernel<<<(BB * NN) / 16 / TW, 32 * TW, 0, stream>>>(
      S, UIh, puw0, ub0, puw1, ub1, lnw, lnb, out);
}
